// dVAE_47115791237431
// MI455X (gfx1250) — matrix operand order repaired and checked by reading, not compiled, not run on hardware
//
#include <hip/hip_runtime.h>
#include <hip/hip_bf16.h>

// ============================================================================
// VQ-VAE forward for MI455X (gfx1250, wave32, WMMA + Tensor Data Mover).
//
// Convs: implicit-GEMM on v_wmma_f32_16x16x32_bf16 (f32 -> bf16 at staging,
// f32 accumulate). Weight tiles are streamed LDS-ward by the TDM
// (tensor_load_to_lds + s_wait_tensorcnt) while the VALU does the im2col
// gather for the activation tile. Codebook search is a WMMA GEMM with a fused
// distance + packed-key global_atomic_min_u64 argmin epilogue.
//
// d_in layout assumption (JAX pytree flatten of {'x': x, 'params': params},
// dict keys sorted alphabetically, tuples (w,b) in order):
//   0: x
//   1,2: conv_final_d   3,4: conv_final_e   5,6: conv_first_d  7,8: conv_first_e
//   9..80: dec groups   81: emb             82..151: enc groups
//
// Workspace requirement: ~170 MB.
// ============================================================================

typedef __attribute__((ext_vector_type(16))) __bf16 v16bf;
typedef __attribute__((ext_vector_type(8)))  float  v8f;
typedef unsigned int u32x4 __attribute__((ext_vector_type(4)));
typedef int          i32x8 __attribute__((ext_vector_type(8)));
typedef int          i32x4 __attribute__((ext_vector_type(4)));

#define GAINF 0.015625f   // 1/(GROUPS*BPG)^2 = 1/64
#define BETAF 0.25f

#if __has_builtin(__builtin_amdgcn_tensor_load_to_lds) && \
    __has_builtin(__builtin_amdgcn_s_wait_tensorcnt)
#define HAVE_TDM 1
#else
#define HAVE_TDM 0
#endif

__device__ __forceinline__ unsigned ford(float f) {
    unsigned u = __float_as_uint(f);
    return (u & 0x80000000u) ? ~u : (u | 0x80000000u);   // order-preserving map
}

#if HAVE_TDM
// Issue a 2D TDM tile load: tileY rows x tileX elems (f32), row stride
// strideX elems, HW zero-fill past dimX/dimY. Wave-uniform; EXEC ignored.
// clang-23 (amdgpu-toolchain) arity: 6 args (g0, g1, g2, g3, extra, cpol).
__device__ __forceinline__ void tdm_load_2d(unsigned lds_off, const float* g,
                                            unsigned dimX, unsigned dimY,
                                            unsigned tileX, unsigned tileY,
                                            unsigned strideX) {
    unsigned long long ga = (unsigned long long)(uintptr_t)g;
    u32x4 g0;
    g0.x = 1u;                                        // count=1 (valid user D#)
    g0.y = lds_off;                                   // lds_addr (bytes)
    g0.z = (unsigned)(ga & 0xFFFFFFFFull);            // global_addr[31:0]
    g0.w = (unsigned)((ga >> 32) & 0x01FFFFFFull) | (2u << 30);  // [56:32]|type=2
    i32x8 g1;
    g1[0] = (int)(2u << 16);                          // data_size=2 (4 bytes)
    g1[1] = (int)((dimX & 0xFFFFu) << 16);            // tensor_dim0 lo16 @48
    g1[2] = (int)(((dimX >> 16) & 0xFFFFu) | ((dimY & 0xFFFFu) << 16));
    g1[3] = (int)(((dimY >> 16) & 0xFFFFu) | (tileX << 16));   // tile_dim0 @112
    g1[4] = (int)(tileY & 0xFFFFu);                   // tile_dim1 @128
    g1[5] = (int)strideX;                             // tensor_dim0_stride lo32
    g1[6] = 0;
    g1[7] = 0;
    i32x4 z4 = {0, 0, 0, 0};
    i32x8 z8 = {0, 0, 0, 0, 0, 0, 0, 0};
    __builtin_amdgcn_tensor_load_to_lds(g0, g1, z4, z4, z8, 0);
}
#endif

// Build a bf16 A-fragment from a raw f32 LDS tile (row m, K-block kb):
// elements 0..15 <- k = kb+16h..kb+16h+15 (contiguous).
__device__ __forceinline__ v16bf afrag_raw(const float* T, int m, int kb, int half) {
    const float* p1 = T + m * 64 + kb + 16 * half;
    const float* p2 = p1 + 8;
    v16bf r;
#pragma unroll
    for (int e = 0; e < 8; ++e) r[e] = (__bf16)p1[e];
#pragma unroll
    for (int e = 0; e < 8; ++e) r[8 + e] = (__bf16)p2[e];
    return r;
}

// bf16 B-fragment from raw f32 LDS tile (row n, K-block kb): k = kb+16h+e.
__device__ __forceinline__ v16bf bfrag_raw(const float* T, int n, int kb, int half) {
    const float* p = T + n * 64 + kb + 16 * half;
    v16bf r;
#pragma unroll
    for (int e = 0; e < 16; ++e) r[e] = (__bf16)p[e];
    return r;
}

// ---------------------------------------------------------------------------
// Generic implicit-GEMM convolution, stride 1, SAME padding, odd kernel.
// Block: 128 threads = 4 wave32; tile 64(Cout) x 64(pixels); K-step 64.
// Weight tile -> LDS via TDM (raw f32); activation tile -> LDS via VALU
// im2col gather, pre-permuted into the 16-bit WMMA B layout.
// ---------------------------------------------------------------------------
__global__ __launch_bounds__(128)
void k_conv_wmma(const float* __restrict__ in, const float* __restrict__ w,
                 const float* __restrict__ bias, const float* __restrict__ ident,
                 float* __restrict__ out,
                 int N, int Cin, int Cout, int H, int W, int Kw,
                 int reluIn, float gain)
{
    __shared__ __align__(32) float  Wraw[64 * 64];   // 16 KB raw f32 weights
    __shared__ __align__(32) __bf16 Bs[2 * 2048];    //  8 KB permuted bf16

    const int tid  = threadIdx.x;
    const int lane = tid & 31;
    const int wid  = tid >> 5;
    const int l15  = lane & 15;
    const int half = lane >> 4;

    const int HW   = H * W;
    const int KK   = Kw * Kw;
    const int Ktot = Cin * KK;
    const int pad  = Kw >> 1;
    const int tileM = blockIdx.x * 64;
    const int tileN = blockIdx.y * 64;
    const int totalPix = N * HW;

    // per-thread fixed pixel for B staging (n = tid & 63), decomposed once
    const int n   = tid & 63;
    const int u_t = tid >> 6;              // which 32-wide K half this thread fills
    const int p   = tileN + n;
    const bool pix_ok = (p < totalPix);
    int nb = 0, y = 0, x = 0;
    if (pix_ok) {
        nb = p / HW; int yx = p - nb * HW;
        y = yx / W;  x = yx - y * W;
    }

    v8f acc[4] = {};

    for (int k0 = 0; k0 < Ktot; k0 += 64) {
        // ---- A tile: TDM streams 64 x 64 f32 weights into LDS ----
#if HAVE_TDM
        if (tid < 32) {
            tdm_load_2d((unsigned)(uintptr_t)Wraw,
                        w + (size_t)tileM * Ktot + k0,
                        (unsigned)(Ktot - k0), (unsigned)(Cout - tileM),
                        64u, 64u, (unsigned)Ktot);
            __builtin_amdgcn_s_wait_tensorcnt(0);
        }
#else
        for (int i = 0; i < 32; ++i) {
            int idx = tid + i * 128;
            int mm = idx >> 6, kk = idx & 63;
            int co = tileM + mm, kg = k0 + kk;
            Wraw[idx] = (co < Cout && kg < Ktot) ? w[(size_t)co * Ktot + kg] : 0.f;
        }
#endif
        // ---- B tile: im2col gather (32 contiguous k for fixed pixel) ----
        {
            int kg = k0 + u_t * 32;
            int ci = kg / KK;  int r = kg - ci * KK;
            int ky = r / Kw;   int kx = r - ky * Kw;
            for (int j = 0; j < 32; ++j) {
                float v = 0.f;
                if (pix_ok && (kg + j) < Ktot) {
                    int iy = y + ky - pad, ix = x + kx - pad;
                    if ((unsigned)iy < (unsigned)H && (unsigned)ix < (unsigned)W) {
                        v = in[((size_t)(nb * Cin + ci)) * HW + iy * W + ix];
                        if (reluIn) v = fmaxf(v, 0.f);
                    }
                }
                Bs[(u_t << 11) + (((n * 2 + (j >> 4)) << 4) | (j & 15))] = (__bf16)v;
                if (++kx == Kw) { kx = 0; if (++ky == Kw) { ky = 0; ++ci; } }
            }
        }
        __syncthreads();

#pragma unroll
        for (int u = 0; u < 2; ++u) {
            v16bf af = afrag_raw(Wraw, wid * 16 + l15, u * 32, half);
#pragma unroll
            for (int s = 0; s < 4; ++s) {
                v16bf bf = *(const v16bf*)(Bs + (u << 11) +
                                           (((s * 16 + l15) * 2 + half) << 4));
                acc[s] = __builtin_amdgcn_wmma_f32_16x16x32_bf16(
                    false, af, false, bf, (short)0, acc[s], false, false);
            }
        }
        __syncthreads();
    }

    // ---- epilogue: bias (+ fused residual), scattered f32 stores ----
#pragma unroll
    for (int s = 0; s < 4; ++s) {
        int pp = tileN + s * 16 + l15;
        int nb2 = pp / HW, yx2 = pp - nb2 * HW;
#pragma unroll
        for (int r = 0; r < 8; ++r) {
            int co = tileM + wid * 16 + r + 8 * half;
            if (co < Cout && pp < totalPix) {
                size_t oi = ((size_t)(nb2 * Cout + co)) * HW + yx2;
                float v = acc[s][r] + bias[co];
                if (ident) v = ident[oi] + gain * v;
                out[oi] = v;
            }
        }
    }
}

// ---------------------------------------------------------------------------
// VQ distance GEMM: G[t][v] = xf[t,:] . emb[v,:],  T=1024, V=8192, K=256.
// emb tile -> LDS via TDM; xf tile gathered from z_e (NCHW 4x256x16x16).
// Fused epilogue: d = |x|^2+|e|^2-2dot, argmin via atomic_min_u64 on
// (ordered_float << 32) | index  (ties -> lowest index, like jnp.argmin).
// ---------------------------------------------------------------------------
__global__ __launch_bounds__(128)
void k_dist_wmma(const float* __restrict__ ze, const float* __restrict__ emb,
                 const float* __restrict__ sx, const float* __restrict__ se,
                 unsigned long long* __restrict__ amin)
{
    __shared__ __align__(32) float  Braw[64 * 64];   // 16 KB raw f32 emb tile
    __shared__ __align__(32) __bf16 As[2 * 2048];    //  8 KB permuted bf16 xf

    const int tid = threadIdx.x, lane = tid & 31, wid = tid >> 5;
    const int l15 = lane & 15, half = lane >> 4;
    const int tileT = blockIdx.y * 64, tileV = blockIdx.x * 64;

    const int m   = tid & 63;              // fixed t-row this thread stages
    const int u_t = tid >> 6;
    const int t   = tileT + m;
    const int nb  = t >> 8, rem = t & 255; // 16x16 spatial per batch image

    v8f acc[4] = {};

    for (int k0 = 0; k0 < 256; k0 += 64) {
#if HAVE_TDM
        if (tid < 32) {
            tdm_load_2d((unsigned)(uintptr_t)Braw,
                        emb + (size_t)tileV * 256 + k0,
                        (unsigned)(256 - k0), (unsigned)(8192 - tileV),
                        64u, 64u, 256u);
            __builtin_amdgcn_s_wait_tensorcnt(0);
        }
#else
        for (int i = 0; i < 32; ++i) {
            int idx = tid + i * 128;
            Braw[idx] = emb[(size_t)(tileV + (idx >> 6)) * 256 + k0 + (idx & 63)];
        }
#endif
        // xf tile, permuted into the A-fragment layout (32 k per thread)
        for (int j = 0; j < 32; ++j) {
            int kg = k0 + u_t * 32 + j;
            float v = ze[((size_t)(nb * 256 + kg)) * 256 + rem];
            int kk = j & 15, h = j >> 4, e = kk;
            As[(u_t << 11) + ((m * 2 + h) << 4) + e] = (__bf16)v;
        }
        __syncthreads();

#pragma unroll
        for (int u = 0; u < 2; ++u) {
            v16bf af = *(const v16bf*)(As + (u << 11) +
                                       (((wid * 16 + l15) * 2 + half) << 4));
#pragma unroll
            for (int s = 0; s < 4; ++s) {
                v16bf bf = bfrag_raw(Braw, s * 16 + l15, u * 32, half);
                acc[s] = __builtin_amdgcn_wmma_f32_16x16x32_bf16(
                    false, af, false, bf, (short)0, acc[s], false, false);
            }
        }
        __syncthreads();
    }

#pragma unroll
    for (int s = 0; s < 4; ++s) {
#pragma unroll
        for (int r = 0; r < 8; ++r) {
            int tt = tileT + wid * 16 + r + 8 * half;
            int vv = tileV + s * 16 + l15;
            float d = sx[tt] + se[vv] - 2.0f * acc[s][r];
            unsigned long long key =
                ((unsigned long long)ford(d) << 32) | (unsigned)vv;
            atomicMin(&amin[tt], key);            // global_atomic_min_u64
        }
    }
}

// ----------------------------- helper kernels ------------------------------
__global__ void k_pool(const float* __restrict__ in, float* __restrict__ out,
                       int NC, int Ho, int Wo) {
    int i = blockIdx.x * blockDim.x + threadIdx.x;
    if (i >= NC * Ho * Wo) return;
    int wo = i % Wo; int tt = i / Wo; int ho = tt % Ho; int nc = tt / Ho;
    int Wi = Wo * 2;
    const float* p = in + (size_t)nc * (4 * Ho * Wo) + (2 * ho) * Wi + 2 * wo;
    out[i] = fmaxf(fmaxf(p[0], p[1]), fmaxf(p[Wi], p[Wi + 1]));
}

__global__ void k_up(const float* __restrict__ in, float* __restrict__ out,
                     int NC, int Ho, int Wo) {
    int i = blockIdx.x * blockDim.x + threadIdx.x;
    if (i >= NC * Ho * Wo) return;
    int wo = i % Wo; int tt = i / Wo; int ho = tt % Ho; int nc = tt / Ho;
    int Wi = Wo / 2;
    out[i] = in[(size_t)nc * (Ho / 2) * Wi + (ho >> 1) * Wi + (wo >> 1)];
}

__global__ void k_sumx2(const float* __restrict__ ze, float* __restrict__ sx) {
    int t = blockIdx.x * blockDim.x + threadIdx.x;
    if (t >= 1024) return;
    int nb = t >> 8, rem = t & 255;
    float s = 0.f;
    for (int e = 0; e < 256; ++e) {
        float v = ze[((size_t)(nb * 256 + e)) * 256 + rem];
        s += v * v;
    }
    sx[t] = s;
}

__global__ void k_sume2(const float* __restrict__ emb, float* __restrict__ se) {
    int v = blockIdx.x * blockDim.x + threadIdx.x;
    if (v >= 8192) return;
    const float* r = emb + (size_t)v * 256;
    float s = 0.f;
    for (int e = 0; e < 256; ++e) s += r[e] * r[e];
    se[v] = s;
}

__global__ void k_init(unsigned long long* amin, int* counts, float* lacc) {
    int i = blockIdx.x * blockDim.x + threadIdx.x;
    if (i < 1024) amin[i] = ~0ull;
    if (i < 8192) counts[i] = 0;
    if (i == 0) lacc[0] = 0.f;
}

__global__ void k_select(const unsigned long long* __restrict__ amin,
                         const float* __restrict__ emb,
                         const float* __restrict__ ze,
                         float* __restrict__ zq,
                         int* __restrict__ counts, float* __restrict__ lacc) {
    int t = blockIdx.x * blockDim.x + threadIdx.x;
    if (t >= 1024) return;
    int idx = (int)(amin[t] & 0xFFFFFFFFull);
    atomicAdd(&counts[idx], 1);
    int nb = t >> 8, rem = t & 255;
    float ss = 0.f;
    for (int e = 0; e < 256; ++e) {
        float q = emb[(size_t)idx * 256 + e];
        float x = ze[((size_t)(nb * 256 + e)) * 256 + rem];
        float d = q - x;
        ss += d * d;
        zq[((size_t)(nb * 256 + e)) * 256 + rem] = q;  // straight-through fwd
    }
    atomicAdd(lacc, ss);
}

__global__ void k_finalize(const int* __restrict__ counts,
                           const float* __restrict__ lacc,
                           float* __restrict__ dOut, int outN) {
    __shared__ float sh[256];
    float ent = 0.f;
    for (int v = threadIdx.x; v < 8192; v += 256) {
        float p = (float)counts[v] * (1.0f / 1024.0f);
        ent += p * logf(p + 1e-10f);
    }
    sh[threadIdx.x] = ent;
    __syncthreads();
    for (int s = 128; s > 0; s >>= 1) {
        if (threadIdx.x < s) sh[threadIdx.x] += sh[threadIdx.x + s];
        __syncthreads();
    }
    if (threadIdx.x == 0) {
        float mse = lacc[0] * (1.0f / (1024.0f * 256.0f));
        dOut[0] = (1.0f + BETAF) * mse;
        dOut[outN - 1] = expf(-sh[0]);
    }
}

// ----------------------------- host orchestration ---------------------------
static inline void launch_conv(hipStream_t s, const float* in, const float* w,
                               const float* b, const float* ident, float* out,
                               int N, int Cin, int Cout, int H, int W, int K,
                               int relu, float gain) {
    dim3 grid((unsigned)((Cout + 63) / 64), (unsigned)((N * H * W + 63) / 64));
    k_conv_wmma<<<grid, 128, 0, s>>>(in, w, b, ident, out, N, Cin, Cout, H, W,
                                     K, relu, gain);
}

extern "C" void kernel_launch(void* const* d_in, const int* in_sizes, int n_in,
                              void* d_out, int out_size, void* d_ws,
                              size_t ws_size, hipStream_t stream) {
    auto F = [&](int i) -> const float* { return (const float*)d_in[i]; };

    // ---- workspace carve-out (floats) ----
    float* A0 = (float*)d_ws;             // 16,777,216  (activation ping)
    float* A1 = A0 + 16777216;            // 16,777,216  (activation pong)
    float* T0 = A1 + 16777216;            //  4,194,304  (hidden nh ping)
    float* T1 = T0 + 4194304;             //  4,194,304  (hidden nh pong)
    unsigned long long* AMIN = (unsigned long long*)(T1 + 4194304);  // 1024
    float* SX   = (float*)(AMIN + 1024);  // 1024
    float* SE   = SX + 1024;              // 8192
    int*   CNT  = (int*)(SE + 8192);      // 8192
    float* LACC = (float*)(CNT + 8192);   // 1

    float* X = A0;
    float* Y = A1;

    auto run_block = [&](int c1, int c2, int c3, int c4, int idc, int cin,
                         int cout, int S, int kf, int kl) {
        int nh = cout / 4;
        launch_conv(stream, X, F(c1), F(c1 + 1), nullptr, T0, 4, cin, nh, S, S, kf, 1, 1.f);
        launch_conv(stream, T0, F(c2), F(c2 + 1), nullptr, T1, 4, nh, nh, S, S, 3, 1, 1.f);
        launch_conv(stream, T1, F(c3), F(c3 + 1), nullptr, T0, 4, nh, nh, S, S, 3, 1, 1.f);
        if (idc >= 0) {
            launch_conv(stream, X, F(idc), F(idc + 1), nullptr, Y, 4, cin, cout, S, S, 1, 0, 1.f);
            launch_conv(stream, T0, F(c4), F(c4 + 1), Y, Y, 4, nh, cout, S, S, kl, 1, GAINF);
        } else {
            launch_conv(stream, T0, F(c4), F(c4 + 1), X, Y, 4, nh, cout, S, S, kl, 1, GAINF);
        }
        float* t = X; X = Y; Y = t;
    };

    const int encB[4][2][5] = {
        {{ 82,  84,  86,  88,  -1}, { 90,  92,  94,  96,  -1}},
        {{ 98, 100, 102, 104, 106}, {108, 110, 112, 114, -1}},
        {{116, 118, 120, 122, 124}, {126, 128, 130, 132, -1}},
        {{134, 136, 138, 140, 142}, {144, 146, 148, 150, -1}}};
    const int decB[4][2][5] = {
        {{  9,  11,  13,  15,  17}, { 19,  21,  23,  25,  -1}},
        {{ 27,  29,  31,  33,  35}, { 37,  39,  41,  43,  -1}},
        {{ 45,  47,  49,  51,  53}, { 55,  57,  59,  61,  -1}},
        {{ 63,  65,  67,  69,  71}, { 73,  75,  77,  79,  -1}}};
    const int encOut[4] = {256, 512, 1024, 2048};
    const int decOut[4] = {2048, 1024, 512, 256};

    // ============================= ENCODER =============================
    int S = 128;
    launch_conv(stream, F(0), F(7), F(8), nullptr, X, 4, 3, 256, S, S, 7, 0, 1.f);
    int Cin = 256;
    for (int g = 0; g < 4; ++g) {
        int Co = encOut[g];
        for (int b = 0; b < 2; ++b) {
            const int* bi = encB[g][b];
            run_block(bi[0], bi[1], bi[2], bi[3], bi[4],
                      b == 0 ? Cin : Co, Co, S, /*kf=*/3, /*kl=*/1);
        }
        Cin = Co;
        if (g < 3) {
            int total = 4 * Co * (S / 2) * (S / 2);
            k_pool<<<(total + 255) / 256, 256, 0, stream>>>(X, Y, 4 * Co, S / 2, S / 2);
            float* t = X; X = Y; Y = t;
            S /= 2;
        }
    }
    launch_conv(stream, X, F(3), F(4), nullptr, Y, 4, 2048, 256, 16, 16, 1, 1, 1.f);
    float* ZE = Y;
    float* ZQ = X;

    // ============================ QUANTIZE =============================
    k_sumx2<<<4, 256, 0, stream>>>(ZE, SX);
    k_sume2<<<32, 256, 0, stream>>>(F(81), SE);
    k_init<<<32, 256, 0, stream>>>(AMIN, CNT, LACC);
    k_dist_wmma<<<dim3(128, 16), 128, 0, stream>>>(ZE, F(81), SX, SE, AMIN);
    k_select<<<4, 256, 0, stream>>>(AMIN, F(81), ZE, ZQ, CNT, LACC);
    k_finalize<<<1, 256, 0, stream>>>(CNT, LACC, (float*)d_out, out_size);

    // ============================= DECODER =============================
    X = ZQ; Y = ZE; S = 16;
    launch_conv(stream, X, F(5), F(6), nullptr, Y, 4, 256, 128, S, S, 1, 0, 1.f);
    { float* t = X; X = Y; Y = t; }
    Cin = 128;
    for (int g = 0; g < 4; ++g) {
        int Co = decOut[g];
        for (int b = 0; b < 2; ++b) {
            const int* bi = decB[g][b];
            run_block(bi[0], bi[1], bi[2], bi[3], bi[4],
                      b == 0 ? Cin : Co, Co, S, /*kf=*/1, /*kl=*/3);
        }
        Cin = Co;
        if (g < 3) {
            int total = 4 * Co * (S * 2) * (S * 2);
            k_up<<<(total + 255) / 256, 256, 0, stream>>>(X, Y, 4 * Co, S * 2, S * 2);
            float* t = X; X = Y; Y = t;
            S *= 2;
        }
    }
    launch_conv(stream, X, F(1), F(2), nullptr, (float*)d_out + 1,
                4, 256, 3, 128, 128, 1, 1, 1.f);
}
